// DepletionLSTM_77025943486675
// MI455X (gfx1250) — matrix operand order repaired and checked by reading, not compiled, not run on hardware
//
#include <hip/hip_runtime.h>
#include <cstdint>
#include <cstddef>

typedef unsigned int  u32;
typedef unsigned short u16;

#define Bc   8192
#define Tc   90
#define Fc   7
#define Hc   128
#define SMEM_BYTES 270848   // 2*131072 (weights) + 2*4352 (x,h tiles)

typedef __bf16 bf16x16 __attribute__((ext_vector_type(16)));
typedef float  f32x8   __attribute__((ext_vector_type(8)));

union Frag { uint4 u[2]; bf16x16 v; };

__device__ __forceinline__ u16 f2bf(float x) {
    u32 u = __float_as_uint(x);
    u32 r = (u + 0x7fffu + ((u >> 16) & 1u)) >> 16;
    return (u16)r;
}
__device__ __forceinline__ float sigm(float x)   { return 1.0f / (1.0f + __expf(-x)); }
__device__ __forceinline__ float tanh_f(float x) { return 1.0f - 2.0f / (__expf(2.0f * x) + 1.0f); }

// ---------------------------------------------------------------------------
// Pre-swizzle W [512,128] f32 (row-major, gate-major) into bf16 B-fragment
// order: frag f = kt*32+nt ; within frag: 32 lanes x 8 packed dword pairs.
// Lane l of fragment (kt,nt) holds W^T[k][n] for n = nt*16+(l&15),
// k = kt*32 + 16*(l>>4) + {2v, 2v+1}.
// ---------------------------------------------------------------------------
__global__ __launch_bounds__(256) void swizzle_w(const float* __restrict__ W,
                                                 u32* __restrict__ out) {
    int id   = blockIdx.x * 256 + threadIdx.x;     // 0..32767
    int v    = id & 7;
    int lane = (id >> 3) & 31;
    int f    = id >> 8;
    int kt   = f >> 5, nt = f & 31;
    int n    = nt * 16 + (lane & 15);
    int k    = kt * 32 + ((lane >> 4) << 4) + 2 * v;
    float a = W[n * 128 + k];
    float b = W[n * 128 + k + 1];
    out[id] = (u32)f2bf(a) | ((u32)f2bf(b) << 16);
}

__global__ void bias_sum(const float* __restrict__ a, const float* __restrict__ b,
                         float* __restrict__ o) {
    int i = blockIdx.x * blockDim.x + threadIdx.x;
    if (i < 512) o[i] = a[i] + b[i];
}

// ---------------------------------------------------------------------------
// Input projection (K=7, not worth WMMA) + LayerNorm -> bf16 sequence buffer.
// One 128-thread block per (b,t); thread h owns hidden unit h.
// ---------------------------------------------------------------------------
__global__ __launch_bounds__(128) void inproj(const float* __restrict__ x,
                                              const float* __restrict__ W,
                                              const float* __restrict__ bi,
                                              const float* __restrict__ g,
                                              const float* __restrict__ be,
                                              u16* __restrict__ out) {
    __shared__ float sred[128];
    int h = threadIdx.x;
    size_t bt = blockIdx.x;
    const float* xr = x + bt * Fc;
    float acc = bi[h];
#pragma unroll
    for (int f = 0; f < Fc; ++f) acc += xr[f] * W[h * Fc + f];
    sred[h] = acc; __syncthreads();
    for (int st = 64; st > 0; st >>= 1) { if (h < st) sred[h] += sred[h + st]; __syncthreads(); }
    float mu = sred[0] * (1.0f / 128.0f); __syncthreads();
    float d = acc - mu;
    sred[h] = d * d; __syncthreads();
    for (int st = 64; st > 0; st >>= 1) { if (h < st) sred[h] += sred[h + st]; __syncthreads(); }
    float var = sred[0] * (1.0f / 128.0f);
    float y = d * rsqrtf(var + 1e-5f) * g[h] + be[h];
    out[bt * 128 + h] = f2bf(y);
}

// ---------------------------------------------------------------------------
// Persistent LSTM layer. Each workgroup (256 threads = 8 wave32) owns 16 batch
// rows for all 90 timesteps. Both weight matrices live in LDS in fragment
// order; h lives in LDS (bf16, padded rows); c lives in accumulator-layout
// registers. Wave w computes gate tiles nt = {w, w+8, w+16, w+24} so the cell
// update is register-local per lane. x_{t+1} is prefetched into registers
// while step t's WMMAs run, hiding global latency behind the serial chain.
// ---------------------------------------------------------------------------
__global__ __launch_bounds__(256) void lstm_layer(
    const u16* __restrict__ seqIn, u16* __restrict__ seqOut,
    const uint4* __restrict__ wihF, const uint4* __restrict__ whhF,
    const float* __restrict__ bias, float* __restrict__ lastH,
    int writeSeq, int writeLast)
{
    extern __shared__ char smem[];
    u32* wIH = (u32*)smem;                       // 131072 B, Wih^T fragments
    u32* wHH = (u32*)(smem + 131072);            // 131072 B, Whh^T fragments
    u16* xt  = (u16*)(smem + 262144);            // [16][136] bf16 x tile
    u16* ht  = (u16*)(smem + 262144 + 4352);     // [16][136] bf16 h tile

    const int tid  = threadIdx.x;
    const int w    = tid >> 5;          // wave 0..7 == hidden group
    const int lane = tid & 31;
    const int hi   = lane >> 4;         // high half of wave
    const int lcol = lane & 15;
    const int rowBase = blockIdx.x * 16;

    // Cooperative-copy indices used for x staging / h write-back.
    const int cr = tid >> 4;            // row 0..15
    const int cc = tid & 15;            // 16-byte chunk 0..15
    const uint4* gsrcRow =
        (const uint4*)(seqIn + ((size_t)(rowBase + cr) * Tc) * 128) + cc;

    // Stage weights into LDS once (reused across 90 steps).
    {
        uint4* d0 = (uint4*)wIH;
        uint4* d1 = (uint4*)wHH;
        for (int i = tid; i < 8192; i += 256) { d0[i] = wihF[i]; d1[i] = whhF[i]; }
    }
    // h0 = 0
    for (int i = tid; i < 1088; i += 256) ((u32*)ht)[i] = 0u;

    // Per-lane gate biases (bih+bhh), constant over v within a tile.
    float bgate[4];
#pragma unroll
    for (int gi = 0; gi < 4; ++gi) bgate[gi] = bias[(gi * 8 + w) * 16 + lcol];

    f32x8 cstate = {0.f, 0.f, 0.f, 0.f, 0.f, 0.f, 0.f, 0.f};

    // Prefetch x_0 into registers (consumed at loop top).
    uint4 px = gsrcRow[0];              // t = 0 : offset 0 uint4s
    __syncthreads();

    for (int t = 0; t < Tc; ++t) {
        // --- publish prefetched x_t tile to LDS ---
        *((uint4*)(xt + cr * 136 + cc * 8)) = px;
        __syncthreads();

        // --- issue prefetch of x_{t+1}; latency hidden behind WMMA phase ---
        {
            int tn = (t + 1 < Tc) ? (t + 1) : t;
            px = gsrcRow[(size_t)tn * 16];   // 16 uint4 (=256B) per timestep row
        }

        // --- gates = bias + x_t @ Wih^T + h @ Whh^T (WMMA bf16->f32) ---
        f32x8 acc[4];
#pragma unroll
        for (int gi = 0; gi < 4; ++gi) {
            f32x8 a;
#pragma unroll
            for (int v = 0; v < 8; ++v) a[v] = bgate[gi];
            acc[gi] = a;
        }

#pragma unroll
        for (int src = 0; src < 2; ++src) {
            const u16* tile = src ? ht : xt;
            const u32* wf   = src ? wHH : wIH;
#pragma unroll
            for (int kt = 0; kt < 4; ++kt) {
                Frag af;
                const u16* ap = tile + lcol * 136 + kt * 32 + hi * 16;
                af.u[0] = *(const uint4*)(ap);
                af.u[1] = *(const uint4*)(ap + 8);
#pragma unroll
                for (int gi = 0; gi < 4; ++gi) {
                    int nt = gi * 8 + w;
                    const uint4* bp = (const uint4*)(wf + ((kt * 32 + nt) * 32 + lane) * 8);
                    Frag bf;
                    bf.u[0] = bp[0];
                    bf.u[1] = bp[1];
                    acc[gi] = __builtin_amdgcn_wmma_f32_16x16x32_bf16(
                        false, af.v, false, bf.v, (short)0, acc[gi], false, false);
                }
            }
        }
        __syncthreads();   // all reads of ht (and xt) finished before overwrite

        // --- LSTM cell update, fully register-local per lane ---
        float hval[8];
#pragma unroll
        for (int v = 0; v < 8; ++v) {
            float ig = sigm(acc[0][v]);
            float fg = sigm(acc[1][v]);
            float gg = tanh_f(acc[2][v]);
            float og = sigm(acc[3][v]);
            float cv = fg * cstate[v] + ig * gg;
            cstate[v] = cv;
            float hv = og * tanh_f(cv);
            hval[v] = hv;
            ht[(v + 8 * hi) * 136 + (w * 16 + lcol)] = f2bf(hv);
        }
        if (writeLast && t == Tc - 1) {
#pragma unroll
            for (int v = 0; v < 8; ++v)
                lastH[(size_t)(rowBase + v + 8 * hi) * 128 + w * 16 + lcol] = hval[v];
        }
        __syncthreads();   // h tile complete

        if (writeSeq) {
            uint4 val = *(const uint4*)(ht + cr * 136 + cc * 8);
            *((uint4*)(seqOut + ((size_t)(rowBase + cr) * Tc + t) * 128) + cc) = val;
        }
    }
}

// ---------------------------------------------------------------------------
// Head: LayerNorm(last h) -> 128 ReLU -> 64 ReLU -> 30. One 128-thread block
// per batch row; ~0.4 GFLOP total, weights stay L2-resident.
// ---------------------------------------------------------------------------
__global__ __launch_bounds__(128) void head_kernel(
    const float* __restrict__ lastH, const float* __restrict__ g,
    const float* __restrict__ be,
    const float* __restrict__ W1, const float* __restrict__ b1,
    const float* __restrict__ W2, const float* __restrict__ b2,
    const float* __restrict__ W3, const float* __restrict__ b3,
    float* __restrict__ out)
{
    __shared__ float sred[128];
    __shared__ float sln[128];
    __shared__ float sd1[128];
    __shared__ float sd2[64];
    int t = threadIdx.x;
    size_t b = blockIdx.x;
    float x = lastH[b * 128 + t];
    sred[t] = x; __syncthreads();
    for (int st = 64; st > 0; st >>= 1) { if (t < st) sred[t] += sred[t + st]; __syncthreads(); }
    float mu = sred[0] * (1.0f / 128.0f); __syncthreads();
    float d = x - mu;
    sred[t] = d * d; __syncthreads();
    for (int st = 64; st > 0; st >>= 1) { if (t < st) sred[t] += sred[t + st]; __syncthreads(); }
    float var = sred[0] * (1.0f / 128.0f);
    sln[t] = d * rsqrtf(var + 1e-5f) * g[t] + be[t];
    __syncthreads();

    float a1 = b1[t];
    for (int k = 0; k < 128; ++k) a1 += W1[t * 128 + k] * sln[k];
    sd1[t] = fmaxf(a1, 0.0f); __syncthreads();

    if (t < 64) {
        float a2 = b2[t];
        for (int k = 0; k < 128; ++k) a2 += W2[t * 128 + k] * sd1[k];
        sd2[t] = fmaxf(a2, 0.0f);
    }
    __syncthreads();

    if (t < 30) {
        float a3 = b3[t];
        for (int k = 0; k < 64; ++k) a3 += W3[t * 64 + k] * sd2[k];
        out[b * 30 + t] = a3;
    }
}

// ---------------------------------------------------------------------------
extern "C" void kernel_launch(void* const* d_in, const int* in_sizes, int n_in,
                              void* d_out, int out_size, void* d_ws, size_t ws_size,
                              hipStream_t stream)
{
    (void)in_sizes; (void)n_in; (void)out_size; (void)ws_size;

    const float* x     = (const float*)d_in[0];
    const float* W_in  = (const float*)d_in[1];
    const float* b_in  = (const float*)d_in[2];
    const float* g_in  = (const float*)d_in[3];
    const float* be_in = (const float*)d_in[4];
    const float* Wih0  = (const float*)d_in[5];
    const float* Whh0  = (const float*)d_in[6];
    const float* bih0  = (const float*)d_in[7];
    const float* bhh0  = (const float*)d_in[8];
    const float* Wih1  = (const float*)d_in[9];
    const float* Whh1  = (const float*)d_in[10];
    const float* bih1  = (const float*)d_in[11];
    const float* bhh1  = (const float*)d_in[12];
    const float* g_ln  = (const float*)d_in[13];
    const float* be_ln = (const float*)d_in[14];
    const float* W_d1  = (const float*)d_in[15];
    const float* b_d1  = (const float*)d_in[16];
    const float* W_d2  = (const float*)d_in[17];
    const float* b_d2  = (const float*)d_in[18];
    const float* W_d3  = (const float*)d_in[19];
    const float* b_d3  = (const float*)d_in[20];

    char* ws = (char*)d_ws;
    const size_t SEQ = (size_t)Bc * Tc * Hc * 2;   // bf16 sequence buffer bytes
    u16* bufA   = (u16*)ws;                        // LN output / L1 dummy out
    u16* bufB   = (u16*)(ws + SEQ);                // layer0 output sequence
    u32* wihF0  = (u32*)(ws + 2 * SEQ);
    u32* whhF0  = wihF0 + 32768;
    u32* wihF1  = whhF0 + 32768;
    u32* whhF1  = wihF1 + 32768;
    float* bias0 = (float*)(whhF1 + 32768);
    float* bias1 = bias0 + 512;
    float* lastH = bias1 + 512;                    // [B,128] f32

    hipFuncSetAttribute((const void*)lstm_layer,
                        hipFuncAttributeMaxDynamicSharedMemorySize, SMEM_BYTES);

    swizzle_w<<<128, 256, 0, stream>>>(Wih0, wihF0);
    swizzle_w<<<128, 256, 0, stream>>>(Whh0, whhF0);
    swizzle_w<<<128, 256, 0, stream>>>(Wih1, wihF1);
    swizzle_w<<<128, 256, 0, stream>>>(Whh1, whhF1);
    bias_sum<<<2, 256, 0, stream>>>(bih0, bhh0, bias0);
    bias_sum<<<2, 256, 0, stream>>>(bih1, bhh1, bias1);

    inproj<<<Bc * Tc, 128, 0, stream>>>(x, W_in, b_in, g_in, be_in, bufA);

    lstm_layer<<<Bc / 16, 256, SMEM_BYTES, stream>>>(
        bufA, bufB, (const uint4*)wihF0, (const uint4*)whhF0, bias0, lastH, 1, 0);
    lstm_layer<<<Bc / 16, 256, SMEM_BYTES, stream>>>(
        bufB, bufA, (const uint4*)wihF1, (const uint4*)whhF1, bias1, lastH, 0, 1);

    head_kernel<<<Bc, 128, 0, stream>>>(lastH, g_ln, be_ln,
                                        W_d1, b_d1, W_d2, b_d2, W_d3, b_d3,
                                        (float*)d_out);
}
